// DenseEncoderLayer_22668837389006
// MI455X (gfx1250) — matrix operand order repaired and checked by reading, not compiled, not run on hardware
//
#include <hip/hip_runtime.h>
#include <hip/hip_bf16.h>

typedef __attribute__((ext_vector_type(16))) __bf16 v16bf;
typedef __attribute__((ext_vector_type(8)))  float  v8f;

#define DEV __device__ __forceinline__

constexpr int Bsz = 2, Sq = 1024, Dm = 1024, Hh = 16, DHd = 64, Ex = 8, FFd = 4096;
constexpr int MTOK = Bsz * Sq;           // 2048 tokens
constexpr float LNEPS = 1e-5f;

DEV uint4 z4() { uint4 z; z.x = 0u; z.y = 0u; z.z = 0u; z.w = 0u; return z; }

DEV float gelu_f(float x) {
  const float c = 0.7978845608028654f;
  float t = tanhf(c * (x + 0.044715f * x * x * x));
  return 0.5f * x * (1.0f + t);
}

// Low 32 bits of a generic LDS pointer == LDS byte offset (shared aperture
// occupies ADDR[63:32] on CDNA5), which is what ASYNC-to-LDS wants in VDST.
DEV unsigned lds_off(const void* p) { return (unsigned)(unsigned long long)p; }

// CDNA5 async DMA global->LDS, 16B per active lane, tracked by ASYNCcnt.
DEV void async_copy_b128(unsigned lds_byte_off, const void* gsrc) {
  asm volatile("global_load_async_to_lds_b128 %0, %1, off"
               :
               : "v"(lds_byte_off), "v"(gsrc)
               : "memory");
}

DEV void wait_async0() {
#if __has_builtin(__builtin_amdgcn_s_wait_asynccnt)
  __builtin_amdgcn_s_wait_asynccnt(0);
#else
  asm volatile("s_wait_asynccnt 0x0" ::: "memory");
#endif
}

// ---------------------------------------------------------------------------
__global__ void k_zero_f32(float* __restrict__ d, long n) {
  long i = (long)blockIdx.x * blockDim.x + threadIdx.x;
  long st = (long)gridDim.x * blockDim.x;
  for (; i < n; i += st) d[i] = 0.0f;
}

// ---------------------------------------------------------------------------
// Tiled transpose + fp32->bf16 downcast:  src [R][C] -> dst [C][R] (bf16).
// blockIdx.z batches (experts) with element strides sIn/sOut.
// ---------------------------------------------------------------------------
__global__ __launch_bounds__(256) void k_cvtT(
    const float* __restrict__ s, __bf16* __restrict__ d, int R, int C,
    long sIn, long sOut) {
  __shared__ float t[32][33];
  s += (long)blockIdx.z * sIn;
  d += (long)blockIdx.z * sOut;
  int bx = blockIdx.x * 32;            // col tile in src
  int by = blockIdx.y * 32;            // row tile in src
  int tx = threadIdx.x & 31, ty = threadIdx.x >> 5;   // 32 x 8
#pragma unroll
  for (int i = 0; i < 32; i += 8) {
    int rr = by + ty + i, cc = bx + tx;
    t[ty + i][tx] = (rr < R && cc < C) ? s[(long)rr * C + cc] : 0.0f;
  }
  __syncthreads();
#pragma unroll
  for (int i = 0; i < 32; i += 8) {
    int rr = bx + ty + i, cc = by + tx;      // dst is [C][R]
    if (rr < C && cc < R) d[(long)rr * R + cc] = (__bf16)t[tx][ty + i];
  }
}

// ---------------------------------------------------------------------------
// LayerNorm over D=1024, one block per row, 256 threads (4 elems/thread).
// ---------------------------------------------------------------------------
__global__ __launch_bounds__(256) void k_layernorm(
    const float* __restrict__ x, const float* __restrict__ g,
    const float* __restrict__ b, float* __restrict__ yf, __bf16* __restrict__ yb) {
  __shared__ float sm[256];
  int row = blockIdx.x, tid = threadIdx.x;
  const float* xr = x + (long)row * Dm;
  float v[4];
#pragma unroll
  for (int i = 0; i < 4; i++) v[i] = xr[tid + i * 256];
  float s = v[0] + v[1] + v[2] + v[3];
  sm[tid] = s; __syncthreads();
  for (int o = 128; o > 0; o >>= 1) { if (tid < o) sm[tid] += sm[tid + o]; __syncthreads(); }
  float mu = sm[0] * (1.0f / Dm); __syncthreads();
  float ss = 0.0f;
#pragma unroll
  for (int i = 0; i < 4; i++) { float dd = v[i] - mu; ss += dd * dd; }
  sm[tid] = ss; __syncthreads();
  for (int o = 128; o > 0; o >>= 1) { if (tid < o) sm[tid] += sm[tid + o]; __syncthreads(); }
  float inv = rsqrtf(sm[0] * (1.0f / Dm) + LNEPS);
#pragma unroll
  for (int i = 0; i < 4; i++) {
    int c = tid + i * 256;
    float y = (v[i] - mu) * inv * g[c] + b[c];
    if (yf) yf[(long)row * Dm + c] = y;
    if (yb) yb[(long)row * Dm + c] = (__bf16)y;
  }
}

// ---------------------------------------------------------------------------
// Double-buffered async WMMA bf16 GEMM:
//   C[M,N] = A[M,K] (row-major) * B stored as B^T, i.e. [N][K] row-major.
// BM=BN=128, BK=32, 256 threads = 8 waves in 4(m) x 2(n); each wave 2x4
// tiles of v_wmma_f32_16x16x32_bf16. Staging: global_load_async_to_lds_b128
// into ping-pong LDS buffers, s_wait_asynccnt + one barrier per k-step.
// MODE 0: C(fp32) = acc
// MODE 1: Cb(bf16) = gelu(acc + bias[n])
// MODE 2: C(fp32) += gate[m*gstride+gidx] * (acc + bias[n])
// ---------------------------------------------------------------------------
template <int MODE>
__global__ __launch_bounds__(256) void k_gemm(
    const __bf16* __restrict__ A, const __bf16* __restrict__ Bm,
    float* __restrict__ C, __bf16* __restrict__ Cb,
    int Md, int Nd, int Kd, int lda, int ldb, int ldc,
    long sA, long sB, long sC,
    const float* __restrict__ bias,
    const float* __restrict__ gate, int gstride, int gidx) {
  __shared__ __bf16 As[2][128 * 40];   // row stride 40 elems keeps 16B alignment
  __shared__ __bf16 Bs[2][128 * 40];   // [n][k]

  int tid = threadIdx.x;
  int bm = blockIdx.y * 128, bn = blockIdx.x * 128;
  long bz = blockIdx.z;
  A += bz * sA; Bm += bz * sB;

  int wave = tid >> 5, lane = tid & 31;
  int wm = wave >> 1, wn = wave & 1;
  int half = lane >> 4, l15 = lane & 15;

  // per-thread staging slot: row r of the 128-row tile, 16-elem segment
  int r = tid >> 1, seg = (tid & 1) * 16;
  const __bf16* Ag = A + (long)(bm + r) * lda + seg;
  const __bf16* Bg = Bm + (long)(bn + r) * ldb + seg;
  bool aok = (bm + r) < Md;
  bool bok = (bn + r) < Nd;
  unsigned lA[2], lB[2];
  lA[0] = lds_off(&As[0][r * 40 + seg]);
  lA[1] = lds_off(&As[1][r * 40 + seg]);
  lB[0] = lds_off(&Bs[0][r * 40 + seg]);
  lB[1] = lds_off(&Bs[1][r * 40 + seg]);

  // rows never written by async copies (M/N tails) -> zero once
  if (!aok) {
#pragma unroll
    for (int bb = 0; bb < 2; bb++) {
      *(uint4*)&As[bb][r * 40 + seg] = z4();
      *(uint4*)&As[bb][r * 40 + seg + 8] = z4();
    }
  }
  if (!bok) {
#pragma unroll
    for (int bb = 0; bb < 2; bb++) {
      *(uint4*)&Bs[bb][r * 40 + seg] = z4();
      *(uint4*)&Bs[bb][r * 40 + seg + 8] = z4();
    }
  }

  v8f acc[2][4];
#pragma unroll
  for (int i = 0; i < 2; i++)
#pragma unroll
    for (int j = 0; j < 4; j++) acc[i][j] = (v8f)(0.0f);

  auto stage = [&](int buf, int k0) {
    if (aok) {
      async_copy_b128(lA[buf], Ag + k0);
      async_copy_b128(lA[buf] + 16, Ag + k0 + 8);
    }
    if (bok) {
      async_copy_b128(lB[buf], Bg + k0);
      async_copy_b128(lB[buf] + 16, Bg + k0 + 8);
    }
  };

  stage(0, 0);
  for (int k0 = 0; k0 < Kd; k0 += 32) {
    int buf = (k0 >> 5) & 1;
    wait_async0();        // our previous-stage DMAs have landed in LDS
    __syncthreads();      // everyone's have; also fences buffer reuse
    if (k0 + 32 < Kd) stage(buf ^ 1, k0 + 32);

    union Fr { uint4 u[2]; v16bf v; };
    Fr af[2], bf[4];
#pragma unroll
    for (int i = 0; i < 2; i++) {   // A 16x32: lane->row l15; K = {8h..8h+7, 16+8h..}
      const __bf16* p = &As[buf][(wm * 32 + i * 16 + l15) * 40 + 8 * half];
      af[i].u[0] = *(const uint4*)p;
      af[i].u[1] = *(const uint4*)(p + 16);
    }
#pragma unroll
    for (int j = 0; j < 4; j++) {   // B 32x16: lane->col l15; K = {8h..8h+7, 16+8h..}
      const __bf16* p = &Bs[buf][(wn * 64 + j * 16 + l15) * 40 + 8 * half];
      bf[j].u[0] = *(const uint4*)p;
      bf[j].u[1] = *(const uint4*)(p + 16);
    }
#pragma unroll
    for (int i = 0; i < 2; i++)
#pragma unroll
      for (int j = 0; j < 4; j++)
        acc[i][j] = __builtin_amdgcn_wmma_f32_16x16x32_bf16(
            false, af[i].v, false, bf[j].v, (short)0, acc[i][j], false, false);
  }

  // epilogue: C/D layout: lane -> n = l15, m = vgpr + 8*half
  float* Cp = C ? C + bz * sC : nullptr;
  __bf16* Cbp = Cb ? Cb + bz * sC : nullptr;
#pragma unroll
  for (int i = 0; i < 2; i++) {
    int gm0 = bm + wm * 32 + i * 16 + half * 8;
#pragma unroll
    for (int j = 0; j < 4; j++) {
      int gn = bn + wn * 64 + j * 16 + l15;
      if (gn < Nd) {
        float bvl = bias ? bias[gn] : 0.0f;
#pragma unroll
        for (int rr = 0; rr < 8; rr++) {
          int gm = gm0 + rr;
          if (gm < Md) {
            float v = acc[i][j][rr];
            if (MODE == 0) {
              Cp[(long)gm * ldc + gn] = v;
            } else if (MODE == 1) {
              Cbp[(long)gm * ldc + gn] = (__bf16)gelu_f(v + bvl);
            } else {
              float gt = gate[(long)gm * gstride + gidx];
              Cp[(long)gm * ldc + gn] += gt * (v + bvl);
            }
          }
        }
      }
    }
  }
}

// ---------------------------------------------------------------------------
// QK l2-norm + scale; q,k -> [b,h,s,d] bf16 (QK_SCALE folded into q),
// v -> [b,h,d,s] bf16 (pre-transposed for the P*V GEMM's B^T operand).
// ---------------------------------------------------------------------------
__global__ __launch_bounds__(64) void k_qknorm(
    const float* __restrict__ qf, const float* __restrict__ kf, const float* __restrict__ vf,
    const float* __restrict__ qs, const float* __restrict__ ks,
    __bf16* __restrict__ qb, __bf16* __restrict__ kb, __bf16* __restrict__ vbT) {
  __shared__ float s1[64], s2[64];
  int tok = blockIdx.x, h = blockIdx.y, d = threadIdx.x;
  int b = tok / Sq, s = tok % Sq;
  long src = (long)tok * Dm + h * DHd + d;
  float q = qf[src], k = kf[src], v = vf[src];
  s1[d] = q * q; s2[d] = k * k; __syncthreads();
  for (int o = 32; o > 0; o >>= 1) {
    if (d < o) { s1[d] += s1[d + o]; s2[d] += s2[d + o]; }
    __syncthreads();
  }
  float qn = q * rsqrtf(s1[0] + 1e-12f) * qs[h * DHd + d] * 10.0f;  // QK_SCALE
  float kn = k * rsqrtf(s2[0] + 1e-12f) * ks[h * DHd + d];
  long bh = (long)b * Hh + h;
  qb[(bh * Sq + s) * DHd + d] = (__bf16)qn;
  kb[(bh * Sq + s) * DHd + d] = (__bf16)kn;
  vbT[(bh * DHd + d) * Sq + s] = (__bf16)v;
}

// ---------------------------------------------------------------------------
// Causal softmax per row; writes probabilities as bf16 for the P*V WMMA GEMM.
// ---------------------------------------------------------------------------
__global__ __launch_bounds__(256) void k_causal_softmax(
    const float* __restrict__ logits, __bf16* __restrict__ P) {
  __shared__ float sm[256];
  int row = blockIdx.x, tid = threadIdx.x;
  long bh = blockIdx.y;
  const float* L = logits + (bh * Sq + row) * (long)Sq;
  __bf16* Pr = P + (bh * Sq + row) * (long)Sq;
  int n = row + 1;
  float mx = -3.4e38f;
  for (int j = tid; j < n; j += 256) mx = fmaxf(mx, L[j]);
  sm[tid] = mx; __syncthreads();
  for (int o = 128; o > 0; o >>= 1) { if (tid < o) sm[tid] = fmaxf(sm[tid], sm[tid + o]); __syncthreads(); }
  mx = sm[0]; __syncthreads();
  float s = 0.0f;
  for (int j = tid; j < n; j += 256) s += __expf(L[j] - mx);
  sm[tid] = s; __syncthreads();
  for (int o = 128; o > 0; o >>= 1) { if (tid < o) sm[tid] += sm[tid + o]; __syncthreads(); }
  float inv = 1.0f / sm[0];
  for (int j = tid; j < Sq; j += 256)
    Pr[j] = (__bf16)((j < n) ? __expf(L[j] - mx) * inv : 0.0f);
}

// [b,h,s,d] fp32 -> [b,s,h*d] bf16
__global__ void k_permute_attnv(const float* __restrict__ a, __bf16* __restrict__ o) {
  long N = (long)MTOK * Dm;
  long i = (long)blockIdx.x * blockDim.x + threadIdx.x;
  long st = (long)gridDim.x * blockDim.x;
  for (; i < N; i += st) {
    long tok = i >> 10;
    int c = (int)(i & 1023);
    int h = c >> 6, d = c & 63;
    long b = tok / Sq, s = tok % Sq;
    o[i] = (__bf16)a[(((b * Hh + h) * Sq) + s) * DHd + d];
  }
}

// gate = softmax(h2 @ gate_w + gate_b) over E=8 experts, one block per token
__global__ __launch_bounds__(256) void k_gate(
    const float* __restrict__ h2, const float* __restrict__ gw,
    const float* __restrict__ gbv, float* __restrict__ gate) {
  __shared__ float sm[256];
  __shared__ float gl[Ex];
  int row = blockIdx.x, tid = threadIdx.x;
  const float* hr = h2 + (long)row * Dm;
  float hv[4];
#pragma unroll
  for (int i = 0; i < 4; i++) hv[i] = hr[tid + i * 256];
  for (int e = 0; e < Ex; e++) {
    float p = 0.0f;
#pragma unroll
    for (int i = 0; i < 4; i++) p += hv[i] * gw[(long)(tid + i * 256) * Ex + e];
    sm[tid] = p; __syncthreads();
    for (int o = 128; o > 0; o >>= 1) { if (tid < o) sm[tid] += sm[tid + o]; __syncthreads(); }
    if (tid == 0) gl[e] = sm[0] + gbv[e];
    __syncthreads();
  }
  if (tid == 0) {
    float mx = gl[0];
    for (int e = 1; e < Ex; e++) mx = fmaxf(mx, gl[e]);
    float s = 0.0f;
    for (int e = 0; e < Ex; e++) { gl[e] = __expf(gl[e] - mx); s += gl[e]; }
    float inv = 1.0f / s;
    for (int e = 0; e < Ex; e++) gate[(long)row * Ex + e] = gl[e] * inv;
  }
}

// ---------------------------------------------------------------------------
extern "C" void kernel_launch(void* const* d_in, const int* in_sizes, int n_in,
                              void* d_out, int out_size, void* d_ws, size_t ws_size,
                              hipStream_t stream) {
  (void)in_sizes; (void)n_in; (void)out_size; (void)ws_size;
  const float* x      = (const float*)d_in[0];
  const float* ln_g   = (const float*)d_in[1];
  const float* ln_b   = (const float*)d_in[2];
  const float* Wq     = (const float*)d_in[3];
  const float* Wk     = (const float*)d_in[4];
  const float* Wv     = (const float*)d_in[5];
  const float* qscale = (const float*)d_in[6];
  const float* kscale = (const float*)d_in[7];
  const float* Wo     = (const float*)d_in[8];
  const float* gate_w = (const float*)d_in[9];
  const float* gate_bv= (const float*)d_in[10];
  const float* w1     = (const float*)d_in[11];
  const float* b1     = (const float*)d_in[12];
  const float* w2     = (const float*)d_in[13];
  const float* b2     = (const float*)d_in[14];
  float* out = (float*)d_out;

  char* W = (char*)d_ws;
  size_t off = 0;
  auto take = [&](size_t bytes) -> char* {
    char* p = W + off;
    off = (off + bytes + 255) & ~(size_t)255;
    return p;
  };

  // all weights stored transposed ([N][K]) in bf16
  __bf16* wqT = (__bf16*)take((size_t)Dm * Dm * 2);
  __bf16* wkT = (__bf16*)take((size_t)Dm * Dm * 2);
  __bf16* wvT = (__bf16*)take((size_t)Dm * Dm * 2);
  __bf16* woT = (__bf16*)take((size_t)Dm * Dm * 2);
  __bf16* w1T = (__bf16*)take((size_t)Ex * Dm * FFd * 2);
  __bf16* w2T = (__bf16*)take((size_t)Ex * FFd * Dm * 2);
  __bf16* hb   = (__bf16*)take((size_t)MTOK * Dm * 2);
  float*  qf   = (float*) take((size_t)MTOK * Dm * 4);
  float*  kf   = (float*) take((size_t)MTOK * Dm * 4);
  float*  vf   = (float*) take((size_t)MTOK * Dm * 4);
  __bf16* qb   = (__bf16*)take((size_t)MTOK * Dm * 2);
  __bf16* kb   = (__bf16*)take((size_t)MTOK * Dm * 2);
  __bf16* vbT  = (__bf16*)take((size_t)MTOK * Dm * 2);
  float*  logits = (float*)take((size_t)Bsz * Hh * Sq * Sq * 4);   // 128 MB
  __bf16* Pm   = (__bf16*)take((size_t)Bsz * Hh * Sq * Sq * 2);    // 64 MB
  float*  attnv   = (float*)take((size_t)MTOK * Dm * 4);
  __bf16* attn_o  = (__bf16*)take((size_t)MTOK * Dm * 2);
  float*  attnprj = (float*)take((size_t)MTOK * Dm * 4);
  float*  h2f  = (float*) take((size_t)MTOK * Dm * 4);
  __bf16* h2b  = (__bf16*)take((size_t)MTOK * Dm * 2);
  float*  gate = (float*) take((size_t)MTOK * Ex * 4);
  // logits region is dead after softmax: reuse for MoE scratch
  __bf16* hid_b = (__bf16*)logits;                                  // 16 MB
  float*  moe   = (float*)((char*)logits + ((size_t)1 << 25));      // 8 MB

  // ---- weight downcast + transpose ----
  dim3 tcv(Dm / 32, Dm / 32, 1);
  k_cvtT<<<tcv, 256, 0, stream>>>(Wq, wqT, Dm, Dm, 0, 0);
  k_cvtT<<<tcv, 256, 0, stream>>>(Wk, wkT, Dm, Dm, 0, 0);
  k_cvtT<<<tcv, 256, 0, stream>>>(Wv, wvT, Dm, Dm, 0, 0);
  k_cvtT<<<tcv, 256, 0, stream>>>(Wo, woT, Dm, Dm, 0, 0);
  dim3 tw1(FFd / 32, Dm / 32, Ex);
  k_cvtT<<<tw1, 256, 0, stream>>>(w1, w1T, Dm, FFd, (long)Dm * FFd, (long)Dm * FFd);
  dim3 tw2(Dm / 32, FFd / 32, Ex);
  k_cvtT<<<tw2, 256, 0, stream>>>(w2, w2T, FFd, Dm, (long)FFd * Dm, (long)FFd * Dm);

  // ---- pre-attention LN -> bf16 activations ----
  k_layernorm<<<MTOK, 256, 0, stream>>>(x, ln_g, ln_b, nullptr, hb);

  // ---- QKV projections ----
  dim3 gqkv(Dm / 128, MTOK / 128, 1);
  k_gemm<0><<<gqkv, 256, 0, stream>>>(hb, wqT, qf, nullptr, MTOK, Dm, Dm,
      Dm, Dm, Dm, 0, 0, 0, nullptr, nullptr, 0, 0);
  k_gemm<0><<<gqkv, 256, 0, stream>>>(hb, wkT, kf, nullptr, MTOK, Dm, Dm,
      Dm, Dm, Dm, 0, 0, 0, nullptr, nullptr, 0, 0);
  k_gemm<0><<<gqkv, 256, 0, stream>>>(hb, wvT, vf, nullptr, MTOK, Dm, Dm,
      Dm, Dm, Dm, 0, 0, 0, nullptr, nullptr, 0, 0);

  // ---- l2norm + scales + head permute (v pre-transposed) ----
  k_qknorm<<<dim3(MTOK, Hh), 64, 0, stream>>>(qf, kf, vf, qscale, kscale, qb, kb, vbT);

  // ---- logits = Q @ K^T, batched over b*h = 32 ----
  dim3 glog(Sq / 128, Sq / 128, Bsz * Hh);
  k_gemm<0><<<glog, 256, 0, stream>>>(qb, kb, logits, nullptr, Sq, Sq, DHd,
      DHd, DHd, Sq, (long)Sq * DHd, (long)Sq * DHd, (long)Sq * Sq,
      nullptr, nullptr, 0, 0);

  k_causal_softmax<<<dim3(Sq, Bsz * Hh), 256, 0, stream>>>(logits, Pm);

  // ---- out = P @ V (V already [b,h,d,s] = B^T), batched ----
  dim3 gav(1, Sq / 128, Bsz * Hh);
  k_gemm<0><<<gav, 256, 0, stream>>>(Pm, vbT, attnv, nullptr, Sq, DHd, Sq,
      Sq, Sq, DHd, (long)Sq * Sq, (long)DHd * Sq, (long)Sq * DHd,
      nullptr, nullptr, 0, 0);

  k_permute_attnv<<<2048, 256, 0, stream>>>(attnv, attn_o);

  // ---- output projection ----
  k_gemm<0><<<gqkv, 256, 0, stream>>>(attn_o, woT, attnprj, nullptr,
      MTOK, Dm, Dm, Dm, Dm, Dm, 0, 0, 0, nullptr, nullptr, 0, 0);

  // ---- mid LN (fp32 for gate, bf16 for MoE GEMMs) ----
  k_layernorm<<<MTOK, 256, 0, stream>>>(attnprj, ln_g, ln_b, h2f, h2b);

  // ---- gate softmax ----
  k_gate<<<MTOK, 256, 0, stream>>>(h2f, gate_w, gate_bv, gate);

  // ---- MoE dense combine ----
  k_zero_f32<<<2048, 256, 0, stream>>>(moe, (long)MTOK * Dm);
  for (int e = 0; e < Ex; e++) {
    dim3 g1(FFd / 128, MTOK / 128, 1);
    k_gemm<1><<<g1, 256, 0, stream>>>(h2b, w1T + (long)e * Dm * FFd,
        nullptr, hid_b, MTOK, FFd, Dm, Dm, Dm, FFd, 0, 0, 0,
        b1 + (long)e * FFd, nullptr, 0, 0);
    dim3 g2(Dm / 128, MTOK / 128, 1);
    k_gemm<2><<<g2, 256, 0, stream>>>(hid_b, w2T + (long)e * FFd * Dm,
        moe, nullptr, MTOK, Dm, FFd, FFd, FFd, Dm, 0, 0, 0,
        b2 + (long)e * Dm, gate, Ex, e);
  }

  // ---- final LN -> output ----
  k_layernorm<<<MTOK, 256, 0, stream>>>(moe, ln_g, ln_b, out, nullptr);
}
